// DynamicClustering_12309376270848
// MI455X (gfx1250) — matrix operand order repaired and checked by reading, not compiled, not run on hardware
//
#include <hip/hip_runtime.h>
#include <hip/hip_bf16.h>

typedef __attribute__((ext_vector_type(16))) _Float16 v16h;
typedef __attribute__((ext_vector_type(8)))  float    v8f;
typedef __attribute__((ext_vector_type(2)))  _Float16 v2h;

#define BB 32
#define NN 8192
#define DD 64
#define KC 128
#define ITERS 10

// ---------------------------------------------------------------------------
// WMMA A-fragment loader (wave32, 16-bit 16x32, contiguous k order as in ldB).
// rows m = lane&15; half hi=lane>>4 holds k in [hi*16,hi*16+16)
// ---------------------------------------------------------------------------
__device__ __forceinline__ v16h load_a_frag(const _Float16* X, int kk) {
  int lane = threadIdx.x & 31;
  const _Float16* row = X + (lane & 15) * DD + kk + (lane >> 4) * 16;
  v16h a;
#pragma unroll
  for (int i = 0; i < 8; ++i) a[i] = row[i];
#pragma unroll
  for (int i = 0; i < 8; ++i) a[8 + i] = row[8 + i];
  return a;
}

// Async-stage one batch's centers (KC*DD f16 = 16KB) into LDS.
// 256 threads x 4 iters x 16B lane-transfers; tracked by ASYNCcnt.
__device__ __forceinline__ void stage_centers_async(const _Float16* Cg, _Float16* lds_c, int tid) {
  unsigned lbase = (unsigned)(size_t)lds_c;   // flat addr low 32b == LDS offset
#pragma unroll
  for (int i = 0; i < 4; ++i) {
    int idx = (tid + i * 256) * 8;            // 8 halfs = 16 bytes per transfer
    unsigned lds_addr = lbase + (unsigned)(idx * 2);
    unsigned long long ga = (unsigned long long)(const void*)(Cg + idx);
    asm volatile("global_load_async_to_lds_b128 %0, %1, off"
                 :: "v"(lds_addr), "v"(ga) : "memory");
  }
  asm volatile("s_wait_asynccnt 0" ::: "memory");
}

// ---------------------------------------------------------------------------
// LayerNorm: one wave per point, 2 elements per lane, shuffle reduction.
// ---------------------------------------------------------------------------
__global__ __launch_bounds__(256) void ln_kernel(const float* __restrict__ patches,
                                                 const float* __restrict__ gamma,
                                                 const float* __restrict__ beta,
                                                 _Float16* __restrict__ xn) {
  int gwave = (int)((blockIdx.x * blockDim.x + threadIdx.x) >> 5); // point id
  int lane  = threadIdx.x & 31;
  const float2 v = *(const float2*)(patches + (size_t)gwave * DD + lane * 2);
  float s  = v.x + v.y;
  float sq = v.x * v.x + v.y * v.y;
#pragma unroll
  for (int off = 16; off >= 1; off >>= 1) {
    s  += __shfl_xor(s,  off, 32);
    sq += __shfl_xor(sq, off, 32);
  }
  float mean = s * (1.0f / DD);
  float var  = sq * (1.0f / DD) - mean * mean;
  float rstd = rsqrtf(var + 1e-5f);
  float y0 = (v.x - mean) * rstd * gamma[lane * 2]     + beta[lane * 2];
  float y1 = (v.y - mean) * rstd * gamma[lane * 2 + 1] + beta[lane * 2 + 1];
  v2h h; h[0] = (_Float16)y0; h[1] = (_Float16)y1;
  *(v2h*)(xn + (size_t)gwave * DD + lane * 2) = h;
}

// ---------------------------------------------------------------------------
// centers0 = xn[:, :128, :]; also write f16 copy and c2 = sum(c*c).
// One wave per (b,k).
// ---------------------------------------------------------------------------
__global__ __launch_bounds__(256) void init_centers(const _Float16* __restrict__ xn,
                                                    float* __restrict__ centers,
                                                    _Float16* __restrict__ ch,
                                                    float* __restrict__ c2) {
  int gw   = (int)((blockIdx.x * blockDim.x + threadIdx.x) >> 5); // b*KC + k
  int lane = threadIdx.x & 31;
  int b = gw >> 7, k = gw & (KC - 1);
  const _Float16* src = xn + ((size_t)b * NN + k) * DD + lane * 2;
  float x0 = (float)src[0], x1 = (float)src[1];
  size_t off = (size_t)gw * DD + lane * 2;
  centers[off] = x0; centers[off + 1] = x1;
  v2h h; h[0] = (_Float16)x0; h[1] = (_Float16)x1;
  *(v2h*)(ch + off) = h;
  float sq = x0 * x0 + x1 * x1;
#pragma unroll
  for (int o = 16; o >= 1; o >>= 1) sq += __shfl_xor(sq, o, 32);
  if (lane == 0) c2[gw] = sq;
}

__global__ __launch_bounds__(256) void zero_kernel(float* __restrict__ sums,
                                                   float* __restrict__ counts) {
  int i = blockIdx.x * blockDim.x + threadIdx.x;
  if (i < BB * KC * DD) sums[i] = 0.0f;
  if (i < BB * KC) counts[i] = 0.0f;
}

// ---------------------------------------------------------------------------
// Assignment + accumulation. 8 waves/block, each wave owns a 16-point tile.
// dot computed with WMMA (C = inline 0); argmax of (dot - 0.5*c2) == argmin dist.
// Centers staged into LDS via GLOBAL_LOAD_ASYNC_TO_LDS_B128 once per block.
// Block accumulates sums/counts in LDS, flushes with global f32 atomics.
// ---------------------------------------------------------------------------
__global__ __launch_bounds__(256) void assign_kernel(const _Float16* __restrict__ xn,
                                                     const _Float16* __restrict__ ch,
                                                     const float* __restrict__ c2,
                                                     float* __restrict__ sums,
                                                     float* __restrict__ counts) {
  __shared__ __align__(128) _Float16 lds_c[KC * DD];   // 16 KB staged centers
  __shared__ float lds_sums[KC * DD];                  // 32 KB partial sums
  __shared__ float lds_counts[KC];
  __shared__ int   lds_best[8][16];

  int tid = threadIdx.x, lane = tid & 31, wave = tid >> 5;
  int b  = blockIdx.x >> 6;                      // 64 blocks per batch
  int n0 = ((blockIdx.x & 63) << 7) + wave * 16; // 128 points per block

  // async center staging (helper waits for completion), then LDS-sum zeroing
  stage_centers_async(ch + (size_t)b * KC * DD, lds_c, tid);
  for (int i = tid; i < KC * DD; i += 256) lds_sums[i] = 0.0f;
  if (tid < KC) lds_counts[tid] = 0.0f;
  __syncthreads();

  const _Float16* X   = xn + ((size_t)b * NN + n0) * DD;
  const float*    c2b = c2 + b * KC;
  int nn = lane & 15, hi = lane >> 4;

  v16h a0 = load_a_frag(X, 0);
  v16h a1 = load_a_frag(X, 32);

  // B = C^T fragment from LDS: lane holds column n, contiguous k-chunk of row n
  auto ldB = [&](int t, int kk) {
    const _Float16* row = lds_c + (t * 16 + nn) * DD + kk + hi * 16;
    v16h bf;
#pragma unroll
    for (int i = 0; i < 16; ++i) bf[i] = row[i];
    return bf;
  };

  float bestv[8]; int besti[8];
#pragma unroll
  for (int r = 0; r < 8; ++r) { bestv[r] = -3.402823466e38f; besti[r] = 0; }

#pragma unroll
  for (int t = 0; t < 8; ++t) {
    v16h b0 = ldB(t, 0);
    v16h b1 = ldB(t, 32);
    v8f z = {};
    v8f acc = __builtin_amdgcn_wmma_f32_16x16x32_f16(false, a0, false, b0, (short)0, z,   false, false);
    acc     = __builtin_amdgcn_wmma_f32_16x16x32_f16(false, a1, false, b1, (short)0, acc, false, false);
    float hc2 = 0.5f * c2b[t * 16 + nn];
#pragma unroll
    for (int r = 0; r < 8; ++r) {
      float v = acc[r] - hc2;
      if (v > bestv[r]) { bestv[r] = v; besti[r] = t * 16 + nn; }
    }
  }
  // reduce argmax over the 16 lanes (columns) of each half-wave
#pragma unroll
  for (int r = 0; r < 8; ++r) {
    float v = bestv[r]; int idx = besti[r];
#pragma unroll
    for (int o = 1; o < 16; o <<= 1) {
      float ov = __shfl_xor(v, o, 32);
      int   oi = __shfl_xor(idx, o, 32);
      if (ov > v || (ov == v && oi < idx)) { v = ov; idx = oi; }
    }
    besti[r] = idx;
  }
  if (nn == 0) {
#pragma unroll
    for (int r = 0; r < 8; ++r) lds_best[wave][8 * hi + r] = besti[r];
  }
  __syncthreads();

  // scatter-add: lane adds elements 2*lane, 2*lane+1 of each of its wave's 16 points
#pragma unroll 4
  for (int p = 0; p < 16; ++p) {
    int k = lds_best[wave][p];
    const _Float16* xp = X + p * DD + lane * 2;
    atomicAdd(&lds_sums[k * DD + lane * 2],     (float)xp[0]);
    atomicAdd(&lds_sums[k * DD + lane * 2 + 1], (float)xp[1]);
    if (lane == 0) atomicAdd(&lds_counts[k], 1.0f);
  }
  __syncthreads();

  float* gs = sums + (size_t)b * KC * DD;
  for (int i = tid; i < KC * DD; i += 256) atomicAdd(&gs[i], lds_sums[i]);
  if (tid < KC) atomicAdd(&counts[b * KC + tid], lds_counts[tid]);
}

// ---------------------------------------------------------------------------
// new_c = counts>0 ? sums/max(counts,1) : old_c ; refresh f16 copy and c2.
// One wave per (b,k).
// ---------------------------------------------------------------------------
__global__ __launch_bounds__(256) void update_centers(const float* __restrict__ sums,
                                                      const float* __restrict__ counts,
                                                      float* __restrict__ centers,
                                                      _Float16* __restrict__ ch,
                                                      float* __restrict__ c2) {
  int gw   = (int)((blockIdx.x * blockDim.x + threadIdx.x) >> 5);
  int lane = threadIdx.x & 31;
  float cnt = counts[gw];
  size_t off = (size_t)gw * DD + lane * 2;
  float inv = 1.0f / fmaxf(cnt, 1.0f);
  float n0 = (cnt > 0.0f) ? sums[off] * inv     : centers[off];
  float n1 = (cnt > 0.0f) ? sums[off + 1] * inv : centers[off + 1];
  centers[off] = n0; centers[off + 1] = n1;
  v2h h; h[0] = (_Float16)n0; h[1] = (_Float16)n1;
  *(v2h*)(ch + off) = h;
  float sq = n0 * n0 + n1 * n1;
#pragma unroll
  for (int o = 16; o >= 1; o >>= 1) sq += __shfl_xor(sq, o, 32);
  if (lane == 0) c2[gw] = sq;
}

// ---------------------------------------------------------------------------
// Final pass: soft = softmax over k of 2*(x.c - 0.5*c2)  (x2 term is shift-invariant).
// ---------------------------------------------------------------------------
__global__ __launch_bounds__(256) void final_kernel(const _Float16* __restrict__ xn,
                                                    const _Float16* __restrict__ ch,
                                                    const float* __restrict__ c2,
                                                    float* __restrict__ soft) {
  __shared__ __align__(128) _Float16 lds_c[KC * DD];   // 16 KB staged centers

  int tid = threadIdx.x, lane = tid & 31, wave = tid >> 5;
  int gw = blockIdx.x * 8 + wave;        // 16384 waves, 512 tiles/batch
  int b  = gw >> 9;
  int n0 = (gw & 511) << 4;

  stage_centers_async(ch + (size_t)b * KC * DD, lds_c, tid);
  __syncthreads();

  const _Float16* X   = xn + ((size_t)b * NN + n0) * DD;
  const float*    c2b = c2 + b * KC;
  int nn = lane & 15, hi = lane >> 4;

  v16h a0 = load_a_frag(X, 0);
  v16h a1 = load_a_frag(X, 32);

  auto ldB = [&](int t, int kk) {
    const _Float16* row = lds_c + (t * 16 + nn) * DD + kk + hi * 16;
    v16h bf;
#pragma unroll
    for (int i = 0; i < 16; ++i) bf[i] = row[i];
    return bf;
  };

  float accs[8][8];
#pragma unroll
  for (int t = 0; t < 8; ++t) {
    v16h b0 = ldB(t, 0);
    v16h b1 = ldB(t, 32);
    v8f z = {};
    v8f acc = __builtin_amdgcn_wmma_f32_16x16x32_f16(false, a0, false, b0, (short)0, z,   false, false);
    acc     = __builtin_amdgcn_wmma_f32_16x16x32_f16(false, a1, false, b1, (short)0, acc, false, false);
    float hc2 = 0.5f * c2b[t * 16 + nn];
#pragma unroll
    for (int r = 0; r < 8; ++r) accs[t][r] = acc[r] - hc2;
  }

#pragma unroll
  for (int r = 0; r < 8; ++r) {
    float mx = accs[0][r];
#pragma unroll
    for (int t = 1; t < 8; ++t) mx = fmaxf(mx, accs[t][r]);
#pragma unroll
    for (int o = 1; o < 16; o <<= 1) mx = fmaxf(mx, __shfl_xor(mx, o, 32));
    float se = 0.0f;
#pragma unroll
    for (int t = 0; t < 8; ++t) {
      float e = __expf(2.0f * (accs[t][r] - mx));
      accs[t][r] = e;
      se += e;
    }
#pragma unroll
    for (int o = 1; o < 16; o <<= 1) se += __shfl_xor(se, o, 32);
    float inv = 1.0f / se;
    size_t base = ((size_t)b * NN + n0 + 8 * hi + r) * KC;
#pragma unroll
    for (int t = 0; t < 8; ++t) soft[base + t * 16 + nn] = accs[t][r] * inv;
  }
}

// ---------------------------------------------------------------------------
extern "C" void kernel_launch(void* const* d_in, const int* in_sizes, int n_in,
                              void* d_out, int out_size, void* d_ws, size_t ws_size,
                              hipStream_t stream) {
  const float* patches = (const float*)d_in[0];
  const float* gamma   = (const float*)d_in[1];
  const float* beta    = (const float*)d_in[2];

  float* centers = (float*)d_out;                        // B*K*D f32
  float* soft    = (float*)d_out + (size_t)BB * KC * DD; // B*N*K f32

  char* w = (char*)d_ws;
  _Float16* xn = (_Float16*)w;  w += (size_t)BB * NN * DD * sizeof(_Float16);
  _Float16* ch = (_Float16*)w;  w += (size_t)BB * KC * DD * sizeof(_Float16);
  float* c2    = (float*)w;     w += (size_t)BB * KC * sizeof(float);
  float* sums  = (float*)w;     w += (size_t)BB * KC * DD * sizeof(float);
  float* counts= (float*)w;

  // LayerNorm -> f16 xn  (wave per point)
  ln_kernel<<<(BB * NN) / 8, 256, 0, stream>>>(patches, gamma, beta, xn);
  // centers0 = xn[:, :128, :]
  init_centers<<<(BB * KC) / 8, 256, 0, stream>>>(xn, centers, ch, c2);

  for (int it = 0; it < ITERS; ++it) {
    zero_kernel<<<(BB * KC * DD) / 256, 256, 0, stream>>>(sums, counts);
    assign_kernel<<<BB * (NN / 128), 256, 0, stream>>>(xn, ch, c2, sums, counts);
    update_centers<<<(BB * KC) / 8, 256, 0, stream>>>(sums, counts, centers, ch, c2);
  }

  final_kernel<<<BB * (NN / 16) / 8, 256, 0, stream>>>(xn, ch, c2, soft);
}
